// GNNNet_29815662969020
// MI455X (gfx1250) — matrix operand order repaired and checked by reading, not compiled, not run on hardware
//
#include <hip/hip_runtime.h>
#include <stdint.h>

#define NG    6
#define NNODE 6000
#define NEDGE 96000
#define CDIM  128
#define MT    (NNODE / 16)   // 375 wmma row-tiles, exact (6000 % 16 == 0)

typedef _Float16 half_t;
typedef __attribute__((ext_vector_type(16))) _Float16 v16h;
typedef __attribute__((ext_vector_type(8)))  _Float16 v8h;
typedef __attribute__((ext_vector_type(8)))  float    v8f;

// ---------------------------------------------------------------------------
// WMMA fragment helpers (contiguous k order: element s of lane half hh is k = 16*hh + s), fed by 16-byte loads.
// ---------------------------------------------------------------------------
static __device__ __forceinline__ v16h make_v16(v8h lo, v8h hi) {
  return __builtin_shufflevector(lo, hi, 0, 1, 2, 3, 4, 5, 6, 7,
                                 8, 9, 10, 11, 12, 13, 14, 15);
}

// A 16x32 f16 fragment: lane half hh covers k in {hh*16 .. hh*16+15}
// -> one contiguous 32B chunk per lane.
static __device__ __forceinline__ v16h load_a_frag(const half_t* __restrict__ A,
                                                   int lda, int lane, int kBase) {
  const int m  = lane & 15;
  const int hh = lane >> 4;
  const half_t* p0 = A + (size_t)m * lda + kBase + hh * 16;
  return make_v16(*(const v8h*)p0, *(const v8h*)(p0 + 8));
}

// B 32x16 f16 fragment from n-major storage (element (k,n) at Bt[n*ldb + k]):
// lane half hh covers k in {hh*16 .. hh*16+15} -> one contiguous 32B chunk.
static __device__ __forceinline__ v16h load_bt_frag(const half_t* __restrict__ Bt,
                                                    int ldb, int lane, int kBase) {
  const int n  = lane & 15;
  const int hh = lane >> 4;
  const half_t* p0 = Bt + (size_t)n * ldb + kBase + hh * 16;
  return make_v16(*(const v8h*)p0, *(const v8h*)(p0 + 8));
}

// ---------------------------------------------------------------------------
// WMMA GEMM: C[M x 128] += A[M x 128] (f16 row-major) @ Bt^T, where Bt is the
// 128x128 weight stored n-major. Bt staged once in LDS, read via ds_load_b128.
// Block = 256 threads = 8 waves; wave = 16-row stripe x 8 column tiles.
// ---------------------------------------------------------------------------
__global__ void wmma_gemm_nt(const half_t* __restrict__ A,
                             const half_t* __restrict__ Bt,
                             float* __restrict__ Cm, int mTiles) {
  __shared__ __align__(16) half_t Bs[CDIM * CDIM];
  {
    const uint4* src = (const uint4*)Bt;
    uint4* dst = (uint4*)Bs;
    for (int i = threadIdx.x; i < CDIM * CDIM / 8; i += 256) dst[i] = src[i];
  }
  __syncthreads();

  const int wave = threadIdx.x >> 5;
  const int lane = threadIdx.x & 31;
  const int mTile = blockIdx.x * 8 + wave;
  if (mTile >= mTiles) return;                  // wave-uniform, no later barrier
  const half_t* Abase = A + (size_t)mTile * 16 * CDIM;

  v8f acc[8];
  const v8f vz = {};
#pragma unroll
  for (int j = 0; j < 8; ++j) acc[j] = vz;

  const int n  = lane & 15;
  const int hh = lane >> 4;
  for (int kt = 0; kt < 4; ++kt) {
    const int kBase = kt * 32;
    v16h a = load_a_frag(Abase, CDIM, lane, kBase);
#pragma unroll
    for (int j = 0; j < 8; ++j) {
      const half_t* p = &Bs[(size_t)(j * 16 + n) * CDIM + kBase + hh * 16];
      v16h b = make_v16(*(const v8h*)p, *(const v8h*)(p + 8));
      acc[j] = __builtin_amdgcn_wmma_f32_16x16x32_f16(false, a, false, b,
                                                      (short)0, acc[j], false, false);
    }
  }
  const int mOff = hh * 8;                      // lanes 0-15: M=v, 16-31: M=8+v
#pragma unroll
  for (int j = 0; j < 8; ++j) {
    float* cp = Cm + (size_t)mTile * 16 * CDIM + j * 16 + n;
#pragma unroll
    for (int v = 0; v < 8; ++v)
      cp[(size_t)(mOff + v) * CDIM] += acc[j][v];
  }
}

// ---------------------------------------------------------------------------
// Pairwise similarity: S = xn[i1] @ xn[i2]^T, fused tanh + 21-bin quantize +
// reduce: sim += centers[clip(floor((tanh(s)+1)*21/2),0,20)].
// B panel (128 nodes x 128 k = 32KB) staged in LDS, shared by all 8 waves.
// grid = (ceil(375/8), ceil(375/8), 36), block = 256.
// ---------------------------------------------------------------------------
__global__ void sim_kernel(const half_t* __restrict__ xn, float* __restrict__ sim_acc) {
  __shared__ __align__(16) half_t Bs[CDIM * CDIM];
  __shared__ float rsum[8];

  const int pair = blockIdx.z;
  const int i1 = pair / NG, i2 = pair % NG;
  const half_t* X1 = xn + (size_t)i1 * NNODE * CDIM;
  const half_t* X2 = xn + (size_t)i2 * NNODE * CDIM;

  const int nPanelNode = blockIdx.y * 128;
  {
    const uint4* src = (const uint4*)(X2 + (size_t)nPanelNode * CDIM);
    uint4* dst = (uint4*)Bs;
    for (int i = threadIdx.x; i < CDIM * CDIM / 8; i += 256) {
      const int row = i >> 4;                   // 16 uint4 per 128-half row
      uint4 v;
      if (nPanelNode + row < NNODE) v = src[i];
      else { v.x = 0; v.y = 0; v.z = 0; v.w = 0; }
      dst[i] = v;
    }
  }
  __syncthreads();

  const int wave = threadIdx.x >> 5;
  const int lane = threadIdx.x & 31;
  const int mTile = blockIdx.x * 8 + wave;
  const int nTileBase = blockIdx.y * 8;

  float lsum = 0.0f;
  if (mTile < MT) {                             // wave-uniform
    const half_t* Abase = X1 + (size_t)mTile * 16 * CDIM;
    v8f acc[8];
    const v8f vz = {};
#pragma unroll
    for (int j = 0; j < 8; ++j) acc[j] = vz;

    const int n  = lane & 15;
    const int hh = lane >> 4;
    for (int kt = 0; kt < 4; ++kt) {
      const int kBase = kt * 32;
      v16h a = load_a_frag(Abase, CDIM, lane, kBase);
#pragma unroll
      for (int j = 0; j < 8; ++j) {
        if (nTileBase + j < MT) {               // wave-uniform
          const half_t* p = &Bs[(size_t)(j * 16 + n) * CDIM + kBase + hh * 16];
          v16h b = make_v16(*(const v8h*)p, *(const v8h*)(p + 8));
          acc[j] = __builtin_amdgcn_wmma_f32_16x16x32_f16(false, a, false, b,
                                                          (short)0, acc[j], false, false);
        }
      }
    }
#pragma unroll
    for (int j = 0; j < 8; ++j) {
      if (nTileBase + j < MT) {
#pragma unroll
        for (int v = 0; v < 8; ++v) {
          const float t = tanhf(acc[j][v]);
          float q = floorf((t + 1.0f) * 10.5f);          // * BINS/2
          q = fminf(fmaxf(q, 0.0f), 20.0f);
          lsum += -1.0f + 0.1f * q;                       // centers[q]
        }
      }
    }
  }
  // wave reduce then block reduce (all threads reach the barrier)
  for (int off = 16; off > 0; off >>= 1) lsum += __shfl_xor(lsum, off, 32);
  if (lane == 0) rsum[wave] = lsum;
  __syncthreads();
  if (threadIdx.x == 0) {
    float s = 0.0f;
#pragma unroll
    for (int w = 0; w < 8; ++w) s += rsum[w];
    atomicAdd(&sim_acc[pair], s);
  }
}

// ---------------------------------------------------------------------------
// Support kernels
// ---------------------------------------------------------------------------
__global__ void fill_f32(float* __restrict__ p, long n, float v) {
  long i = (long)blockIdx.x * blockDim.x + threadIdx.x;
  if (i < n) p[i] = v;
}

__global__ void cvt_f16(const float* __restrict__ s, half_t* __restrict__ d, long n) {
  long i = (long)blockIdx.x * blockDim.x + threadIdx.x;
  if (i < n) d[i] = (half_t)s[i];
}

// transpose-convert 4 weight matrices: Wt[l][n][k] = (f16)W[l][k][n]
__global__ void cvt_f16_t(const float* __restrict__ s, half_t* __restrict__ d) {
  long i = (long)blockIdx.x * blockDim.x + threadIdx.x;
  if (i >= 4L * CDIM * CDIM) return;
  const int l = (int)(i >> 14);
  const int r = (int)(i & 16383);
  const int k = r >> 7, n = r & 127;
  d[((size_t)l * CDIM + n) * CDIM + k] = (half_t)s[i];
}

// deg[(g*4+l)*N + col] += edge_attr[g,e,l+2]  (deg pre-filled with 1.0 self loop)
__global__ void deg_accum(const long long* __restrict__ ei, const float* __restrict__ ea,
                          float* __restrict__ deg) {
  long tid = (long)blockIdx.x * blockDim.x + threadIdx.x;
  if (tid >= (long)NG * 4 * NEDGE) return;
  const int g = (int)(tid / (4L * NEDGE));
  const long r = tid % (4L * NEDGE);
  const int l = (int)(r / NEDGE);
  const int e = (int)(r % NEDGE);
  long col = ei[((long)g * 2 + 1) * NEDGE + e];
  col = col < 0 ? 0 : (col >= NNODE ? NNODE - 1 : col);
  const float w = ea[((long)g * NEDGE + e) * 6 + l + 2];
  atomicAdd(&deg[((long)g * 4 + l) * NNODE + col], w);
}

__global__ void deg_to_dis(float* __restrict__ deg, long n) {
  long i = (long)blockIdx.x * blockDim.x + threadIdx.x;
  if (i >= n) return;
  const float v = deg[i];
  deg[i] = (v > 0.0f) ? rsqrtf(fmaxf(v, 1e-30f)) : 0.0f;
}

// one wave-lane handles 4 channels of one edge: gcn[col] += dis[r]*w*dis[c]*h[row]
__global__ void scatter_edges(const float* __restrict__ h_buf, const float* __restrict__ dis,
                              const long long* __restrict__ ei, const float* __restrict__ ea,
                              float* __restrict__ gcn_buf, int g) {
  long tid = (long)blockIdx.x * blockDim.x + threadIdx.x;
  if (tid >= 4L * NEDGE * 32) return;
  const int lane = (int)(tid & 31);
  const long idx = tid >> 5;
  const int l = (int)(idx / NEDGE);
  const int e = (int)(idx % NEDGE);
  long row = ei[((long)g * 2 + 0) * NEDGE + e];
  long col = ei[((long)g * 2 + 1) * NEDGE + e];
  row = row < 0 ? 0 : (row >= NNODE ? NNODE - 1 : row);
  col = col < 0 ? 0 : (col >= NNODE ? NNODE - 1 : col);
  const float w = ea[((long)g * NEDGE + e) * 6 + l + 2];
  const float* dl = dis + ((long)g * 4 + l) * NNODE;
  const float norm = dl[row] * w * dl[col];
  const float* hp = h_buf + ((size_t)l * NNODE + row) * CDIM + lane * 4;
  float* op = gcn_buf + ((size_t)l * NNODE + col) * CDIM + lane * 4;
#pragma unroll
  for (int i = 0; i < 4; ++i) atomicAdd(&op[i], norm * hp[i]);
}

__global__ void self_loops(const float* __restrict__ h_buf, const float* __restrict__ dis,
                           float* __restrict__ gcn_buf, int g) {
  long tid = (long)blockIdx.x * blockDim.x + threadIdx.x;
  if (tid >= 4L * NNODE * 32) return;
  const int lane = (int)(tid & 31);
  const long idx = tid >> 5;
  const int l = (int)(idx / NNODE);
  const int n = (int)(idx % NNODE);
  const float d = dis[((long)g * 4 + l) * NNODE + n];
  const float norm = d * d;
  const float* hp = h_buf + ((size_t)l * NNODE + n) * CDIM + lane * 4;
  float* op = gcn_buf + ((size_t)l * NNODE + n) * CDIM + lane * 4;
#pragma unroll
  for (int i = 0; i < 4; ++i) op[i] += norm * hp[i];
}

__global__ void bias_relu_cvt(const float* __restrict__ gcn, const float* __restrict__ b,
                              half_t* __restrict__ hr) {
  long i = (long)blockIdx.x * blockDim.x + threadIdx.x;
  if (i >= (long)NNODE * CDIM) return;
  const int c = (int)(i & (CDIM - 1));
  hr[i] = (half_t)fmaxf(gcn[i] + b[c], 0.0f);
}

__global__ void mean_kernel(const float* __restrict__ feats, float* __restrict__ out) {
  const int g = blockIdx.x, c = threadIdx.x;
  const float* p = feats + (size_t)g * NNODE * CDIM + c;
  float s = 0.0f;
  for (int n = 0; n < NNODE; ++n) s += p[(size_t)n * CDIM];
  out[g * CDIM + c] = s / (float)NNODE;
}

// one wave per node: L2-normalize 128 channels, emit f16
__global__ void normalize_kernel(const float* __restrict__ feats, half_t* __restrict__ xn) {
  const int lane = threadIdx.x & 31, wave = threadIdx.x >> 5;
  const long node = (long)blockIdx.x * 8 + wave;
  if (node >= (long)NG * NNODE) return;
  const float4 v = *(const float4*)(feats + node * CDIM + lane * 4);
  float ss = v.x * v.x + v.y * v.y + v.z * v.z + v.w * v.w;
  for (int off = 16; off > 0; off >>= 1) ss += __shfl_xor(ss, off, 32);
  const float inv = 1.0f / fmaxf(sqrtf(ss), 1e-12f);
  half_t* q = xn + node * CDIM + lane * 4;
  q[0] = (half_t)(v.x * inv); q[1] = (half_t)(v.y * inv);
  q[2] = (half_t)(v.z * inv); q[3] = (half_t)(v.w * inv);
}

__global__ void finalize_kernel(const float* __restrict__ sim_acc, float* __restrict__ out) {
  const int i = threadIdx.x;
  if (i < NG * NG)
    out[NG * CDIM + i] = sim_acc[i] * (1.0f / ((float)NNODE * (float)NNODE));
}

// ---------------------------------------------------------------------------
static inline int cdiv(long a, long b) { return (int)((a + b - 1) / b); }

extern "C" void kernel_launch(void* const* d_in, const int* in_sizes, int n_in,
                              void* d_out, int out_size, void* d_ws, size_t ws_size,
                              hipStream_t stream) {
  const float*      x  = (const float*)d_in[0];       // [NG, N, 128]
  const long long*  ei = (const long long*)d_in[1];   // [NG, 2, E] int64
  const float*      ea = (const float*)d_in[2];       // [NG, E, 6]
  const float*      gw = (const float*)d_in[3];       // [4, 128, 128]
  const float*      gb = (const float*)d_in[4];       // [4, 128]
  const float*      ws = (const float*)d_in[5];       // [4, 128, 128]
  float* out = (float*)d_out;                         // 768 g_matrix + 36 sim

  char* base = (char*)d_ws;
  size_t off = 0;
  auto alloc = [&](size_t bytes) {
    void* p = base + off;
    off = (off + bytes + 255) & ~(size_t)255;
    return p;
  };
  float*  deg     = (float*)alloc((size_t)NG * 4 * NNODE * 4);        // -> dis in place
  half_t* xh      = (half_t*)alloc((size_t)NG * NNODE * CDIM * 2);
  half_t* Wt      = (half_t*)alloc((size_t)4 * CDIM * CDIM * 2);      // gcn_W transposed
  half_t* Wst     = (half_t*)alloc((size_t)4 * CDIM * CDIM * 2);      // Ws transposed
  float*  h_buf   = (float*)alloc((size_t)4 * NNODE * CDIM * 4);      // per-graph reuse
  float*  gcn_buf = (float*)alloc((size_t)4 * NNODE * CDIM * 4);      // per-graph reuse
  half_t* hr      = (half_t*)alloc((size_t)NNODE * CDIM * 2);         // per-(g,l) reuse
  float*  feats   = (float*)alloc((size_t)NG * NNODE * CDIM * 4);
  half_t* xn      = (half_t*)alloc((size_t)NG * NNODE * CDIM * 2);
  float*  sim_acc = (float*)alloc(NG * NG * 4);
  (void)in_sizes; (void)n_in; (void)out_size; (void)ws_size;

  hipMemsetAsync(feats, 0, (size_t)NG * NNODE * CDIM * 4, stream);
  hipMemsetAsync(sim_acc, 0, NG * NG * 4, stream);

  // degrees (self-loop weight 1.0) -> dis = rsqrt
  const long degN = (long)NG * 4 * NNODE;
  fill_f32<<<cdiv(degN, 256), 256, 0, stream>>>(deg, degN, 1.0f);
  deg_accum<<<cdiv((long)NG * 4 * NEDGE, 256), 256, 0, stream>>>(ei, ea, deg);
  deg_to_dis<<<cdiv(degN, 256), 256, 0, stream>>>(deg, degN);

  // f16 copies for WMMA (weights transposed to n-major for contiguous B frags)
  cvt_f16<<<cdiv((long)NG * NNODE * CDIM, 256), 256, 0, stream>>>(x, xh, (long)NG * NNODE * CDIM);
  cvt_f16_t<<<cdiv((long)4 * CDIM * CDIM, 256), 256, 0, stream>>>(gw, Wt);
  cvt_f16_t<<<cdiv((long)4 * CDIM * CDIM, 256), 256, 0, stream>>>(ws, Wst);

  const int gemmBlocks = cdiv(MT, 8);   // 47
  for (int g = 0; g < NG; ++g) {
    hipMemsetAsync(h_buf, 0, (size_t)4 * NNODE * CDIM * 4, stream);
    for (int l = 0; l < 4; ++l)
      wmma_gemm_nt<<<gemmBlocks, 256, 0, stream>>>(
          xh + (size_t)g * NNODE * CDIM, Wt + (size_t)l * CDIM * CDIM,
          h_buf + (size_t)l * NNODE * CDIM, MT);

    hipMemsetAsync(gcn_buf, 0, (size_t)4 * NNODE * CDIM * 4, stream);
    scatter_edges<<<cdiv(4L * NEDGE * 32, 256), 256, 0, stream>>>(h_buf, deg, ei, ea, gcn_buf, g);
    self_loops<<<cdiv(4L * NNODE * 32, 256), 256, 0, stream>>>(h_buf, deg, gcn_buf, g);

    for (int l = 0; l < 4; ++l) {
      bias_relu_cvt<<<cdiv((long)NNODE * CDIM, 256), 256, 0, stream>>>(
          gcn_buf + (size_t)l * NNODE * CDIM, gb + (size_t)l * CDIM, hr);
      wmma_gemm_nt<<<gemmBlocks, 256, 0, stream>>>(
          hr, Wst + (size_t)l * CDIM * CDIM, feats + (size_t)g * NNODE * CDIM, MT);
    }
  }

  mean_kernel<<<NG, CDIM, 0, stream>>>(feats, out);
  normalize_kernel<<<cdiv((long)NG * NNODE, 8), 256, 0, stream>>>(feats, xn);

  dim3 simGrid(cdiv(MT, 8), cdiv(MT, 8), NG * NG);
  sim_kernel<<<simGrid, 256, 0, stream>>>(xn, sim_acc);

  finalize_kernel<<<1, 64, 0, stream>>>(sim_acc, out);
}
